// Model_39676907883971
// MI455X (gfx1250) — matrix operand order repaired and checked by reading, not compiled, not run on hardware
//
#include <hip/hip_runtime.h>
#include <hip/hip_bf16.h>
#include <stdint.h>

typedef __bf16 bf16_t;
typedef __attribute__((ext_vector_type(16))) __bf16 v16bf;
typedef __attribute__((ext_vector_type(8)))  float  v8f;

#define B_DIM 8
#define SQ 2048
#define SK 2048
#define DH 1024
#define DV 1024
#define DQK 1024

union FragBF { uint4 u[2]; v16bf v; };
union BfPack4 { bf16_t h[4]; uint2 u; };

// A-fragment (16x32 bf16, row-major source, ld elements per row).
// lanes 0-15: row=lane,   K chunks [0..7] and [16..23]
// lanes16-31: row=lane-16,K chunks [8..15] and [24..31]
__device__ inline v16bf load_frag_a(const bf16_t* __restrict base, int ld) {
  int lane = threadIdx.x & 31;
  int m = lane & 15;
  int koff = (lane >> 4) << 3;               // 0 or 8
  const bf16_t* p = base + (size_t)m * ld + koff;
  FragBF f;
  f.u[0] = *(const uint4*)(p);
  f.u[1] = *(const uint4*)(p + 16);
  return f.v;
}

// B-fragment (32x16 bf16) read from a [N x K] row-major source (K contiguous).
// lanes 0-15: col=lane, K [0..7],[16..23] ; lanes 16-31: col=lane-16, K [8..15],[24..31]
__device__ inline v16bf load_frag_b(const bf16_t* __restrict base, int ld) {
  int lane = threadIdx.x & 31;
  int n = lane & 15;
  int koff = (lane >> 4) << 3;               // 0 or 8
  const bf16_t* p = base + (size_t)n * ld + koff;
  FragBF f;
  f.u[0] = *(const uint4*)(p);
  f.u[1] = *(const uint4*)(p + 16);
  return f.v;
}

// ---------------------------------------------------------------------------
// Projection GEMM: C = act(A[MxK] * W[KxN] + bias), A/W f32, C bf16.
// TRANS==1 stores C transposed per batch: out[b][n][s]  (s = row % SK).
// 256 threads, 8 waves: 4x2 wave grid over a 128x128 tile, K-step 64.
// ---------------------------------------------------------------------------
#define KSTEP 64
#define PLD   (KSTEP + 8)   // 72 bf16 -> 144B row stride (16B aligned)

template<int ACT, int TRANS>
__global__ __launch_bounds__(256, 4)
void proj_kernel(const float* __restrict A, const float* __restrict W,
                 const float* __restrict bias, bf16_t* __restrict out,
                 int M, int K, int N) {
  __shared__ alignas(16) bf16_t As[128][PLD];
  __shared__ alignas(16) bf16_t Bs[128][PLD];   // stored as [n][k]
  const int m0 = blockIdx.y * 128;
  const int n0 = blockIdx.x * 128;
  const int t = threadIdx.x;
  const int w = t >> 5;
  const int lane = t & 31;
  const int half = lane >> 4;
  const int col = lane & 15;
  const int wm = w >> 1;     // 0..3  -> 32 rows
  const int wn = w & 1;      // 0..1  -> 64 cols

  v8f acc[2][4];
  for (int i = 0; i < 2; ++i)
    for (int j = 0; j < 4; ++j) { v8f z = {}; acc[i][j] = z; }

  for (int kk = 0; kk < K; kk += KSTEP) {
    // A tile 128x64 f32 -> bf16 LDS (packed 8B stores)
    for (int it = 0; it < 8; ++it) {
      int f = t + it * 256;                 // float4 index, 2048 total
      int row = f >> 4;                     // 16 float4 per row
      int kq = (f & 15) << 2;
      float4 va = *(const float4*)(A + (size_t)(m0 + row) * K + kk + kq);
      BfPack4 p;
      p.h[0] = (bf16_t)va.x; p.h[1] = (bf16_t)va.y;
      p.h[2] = (bf16_t)va.z; p.h[3] = (bf16_t)va.w;
      *(uint2*)&As[row][kq] = p.u;
    }
    // W tile 64x128 f32 -> transposed bf16 LDS [n][k]
    for (int it = 0; it < 8; ++it) {
      int f = t + it * 256;
      int krow = f >> 5;                    // 0..63
      int nq = (f & 31) << 2;
      float4 vb = *(const float4*)(W + (size_t)(kk + krow) * N + n0 + nq);
      Bs[nq + 0][krow] = (bf16_t)vb.x;
      Bs[nq + 1][krow] = (bf16_t)vb.y;
      Bs[nq + 2][krow] = (bf16_t)vb.z;
      Bs[nq + 3][krow] = (bf16_t)vb.w;
    }
    __syncthreads();
    for (int kc = 0; kc < KSTEP; kc += 32) {
      v16bf afr0 = load_frag_a(&As[wm * 32][kc], PLD);
      v16bf afr1 = load_frag_a(&As[wm * 32 + 16][kc], PLD);
      for (int j = 0; j < 4; ++j) {
        v16bf bfr = load_frag_b(&Bs[wn * 64 + 16 * j][kc], PLD);
        acc[0][j] = __builtin_amdgcn_wmma_f32_16x16x32_bf16(
            false, afr0, false, bfr, (short)0, acc[0][j], false, false);
        acc[1][j] = __builtin_amdgcn_wmma_f32_16x16x32_bf16(
            false, afr1, false, bfr, (short)0, acc[1][j], false, false);
      }
    }
    __syncthreads();
  }

  for (int i = 0; i < 2; ++i) {
    for (int j = 0; j < 4; ++j) {
      int n = n0 + wn * 64 + 16 * j + col;
      float bvv = bias[n];
      for (int e = 0; e < 8; ++e) {
        int m = m0 + wm * 32 + 16 * i + 8 * half + e;
        float v = acc[i][j][e] + bvv;
        if (ACT) v = tanhf(v);
        if (!TRANS) {
          out[(size_t)m * N + n] = (bf16_t)v;
        } else {
          int b = m / SK;
          int s = m - b * SK;
          out[((size_t)b * N + n) * SK + s] = (bf16_t)v;
        }
      }
    }
  }
}

// ---------------------------------------------------------------------------
// Fused attention (flash-style): per block = (batch, 32 q rows), 256 threads.
// Loop over 16 k-tiles of 128: S=Q@K^T (WMMA) +mask, online softmax,
// P staged bf16 in LDS, O += P@V with VpT (WMMA).
// launch_bounds min-waves=4 caps the VGPR budget so no VGPR-MSB addressing
// is needed and multiple blocks co-reside per WGP to hide L2 latency.
// ---------------------------------------------------------------------------
#define BQ 32
#define BK 128

__global__ __launch_bounds__(256, 4)
void flash_kernel(const bf16_t* __restrict Qp, const bf16_t* __restrict Kp,
                  const bf16_t* __restrict VpT, const uint8_t* __restrict mask,
                  float* __restrict out) {
  __shared__ alignas(16) bf16_t Ps[BQ][BK + 8];
  __shared__ float rmax_part[4][BQ];
  __shared__ float rsum_part[4][BQ];
  __shared__ float m_state[BQ];
  __shared__ float l_state[BQ];
  __shared__ float scale_s[BQ];

  const int b = blockIdx.y;
  const int q0 = blockIdx.x * BQ;
  const int t = threadIdx.x;
  const int w = t >> 5;
  const int lane = t & 31;
  const int half = lane >> 4;
  const int col = lane & 15;
  const int wq = w >> 2;        // 0..1 : 16 q rows in S-phase
  const int wk = w & 3;         // 0..3 : 32 k cols in S-phase
  const int dv0 = w * 128;      // per-wave DV slice in O-phase

  if (t < BQ) { m_state[t] = -1e30f; l_state[t] = 0.0f; }

  v8f oacc[2][8];
  for (int i = 0; i < 2; ++i)
    for (int j = 0; j < 8; ++j) { v8f z = {}; oacc[i][j] = z; }

  const bf16_t* Qb = Qp + (size_t)b * SQ * DH;
  const bf16_t* Kb = Kp + (size_t)b * SK * DH;
  const bf16_t* Vb = VpT + (size_t)b * DV * SK;
  const uint8_t* mb = mask + (size_t)b * SK;

  __syncthreads();

  for (int k0 = 0; k0 < SK; k0 += BK) {
    // ---- S = Q @ K^T : A,B fragments straight from global (both contiguous)
    v8f s[2];
    for (int j = 0; j < 2; ++j) { v8f z = {}; s[j] = z; }
    for (int dh = 0; dh < DH; dh += 32) {
      v16bf a = load_frag_a(Qb + (size_t)(q0 + wq * 16) * DH + dh, DH);
      for (int j = 0; j < 2; ++j) {
        v16bf bb = load_frag_b(Kb + (size_t)(k0 + wk * 32 + 16 * j) * DH + dh, DH);
        s[j] = __builtin_amdgcn_wmma_f32_16x16x32_bf16(
            false, a, false, bb, (short)0, s[j], false, false);
      }
    }
    // ---- add bool mask (+0/+1 like reference), partial row max
    float rm[8];
    for (int e = 0; e < 8; ++e) rm[e] = -1e30f;
    for (int j = 0; j < 2; ++j) {
      float mv = (float)mb[k0 + wk * 32 + 16 * j + col];
      for (int e = 0; e < 8; ++e) {
        float v = s[j][e] + mv;
        s[j][e] = v;
        rm[e] = fmaxf(rm[e], v);
      }
    }
    for (int off = 8; off >= 1; off >>= 1)
      for (int e = 0; e < 8; ++e)
        rm[e] = fmaxf(rm[e], __shfl_xor(rm[e], off, 32));
    if (col == 0)
      for (int e = 0; e < 8; ++e)
        rmax_part[wk][wq * 16 + 8 * half + e] = rm[e];
    __syncthreads();

    if (t < BQ) {  // online-softmax state update (one thread per row)
      float mt = fmaxf(fmaxf(rmax_part[0][t], rmax_part[1][t]),
                       fmaxf(rmax_part[2][t], rmax_part[3][t]));
      float mo = m_state[t];
      float mn = fmaxf(mo, mt);
      m_state[t] = mn;
      float sc = __expf(mo - mn);
      scale_s[t] = sc;
      l_state[t] *= sc;
    }
    __syncthreads();

    // ---- P = exp(S - m) -> LDS (bf16), partial row sums
    float rs[8];
    for (int e = 0; e < 8; ++e) rs[e] = 0.0f;
    for (int j = 0; j < 2; ++j) {
      for (int e = 0; e < 8; ++e) {
        int row = wq * 16 + 8 * half + e;
        float p = __expf(s[j][e] - m_state[row]);
        rs[e] += p;
        Ps[row][wk * 32 + 16 * j + col] = (bf16_t)p;
      }
    }
    for (int off = 8; off >= 1; off >>= 1)
      for (int e = 0; e < 8; ++e)
        rs[e] += __shfl_xor(rs[e], off, 32);
    if (col == 0)
      for (int e = 0; e < 8; ++e)
        rsum_part[wk][wq * 16 + 8 * half + e] = rs[e];
    __syncthreads();
    if (t < BQ)
      l_state[t] += rsum_part[0][t] + rsum_part[1][t] +
                    rsum_part[2][t] + rsum_part[3][t];

    // ---- rescale O accumulators, then O += P @ V (VpT: K-dim contiguous)
    for (int i = 0; i < 2; ++i) {
      for (int e = 0; e < 8; ++e) {
        float sc = scale_s[i * 16 + 8 * half + e];
        for (int j = 0; j < 8; ++j) oacc[i][j][e] *= sc;
      }
    }
    for (int kc = 0; kc < 4; ++kc) {
      v16bf a[2];
      for (int i = 0; i < 2; ++i)
        a[i] = load_frag_a(&Ps[i * 16][kc * 32], BK + 8);
      for (int j = 0; j < 8; ++j) {
        v16bf bb = load_frag_b(Vb + (size_t)(dv0 + 16 * j) * SK + k0 + kc * 32, SK);
        for (int i = 0; i < 2; ++i)
          oacc[i][j] = __builtin_amdgcn_wmma_f32_16x16x32_bf16(
              false, a[i], false, bb, (short)0, oacc[i][j], false, false);
      }
    }
    __syncthreads();
  }

  // ---- finalize: divide by l, write f32
  for (int i = 0; i < 2; ++i) {
    for (int e = 0; e < 8; ++e) {
      int row = i * 16 + 8 * half + e;
      float inv = 1.0f / l_state[row];
      int q = q0 + row;
      for (int j = 0; j < 8; ++j)
        out[((size_t)b * SQ + q) * DV + dv0 + 16 * j + col] = oacc[i][j][e] * inv;
    }
  }
}

// ---------------------------------------------------------------------------
extern "C" void kernel_launch(void* const* d_in, const int* in_sizes, int n_in,
                              void* d_out, int out_size, void* d_ws, size_t ws_size,
                              hipStream_t stream) {
  const float*   q    = (const float*)d_in[0];
  const float*   k    = (const float*)d_in[1];
  const float*   v    = (const float*)d_in[2];
  const uint8_t* mask = (const uint8_t*)d_in[3];   // jnp.bool_ = 1 byte
  const float*   Wq   = (const float*)d_in[4];
  const float*   bq   = (const float*)d_in[5];
  const float*   Wk   = (const float*)d_in[6];
  const float*   bk   = (const float*)d_in[7];
  const float*   Wv   = (const float*)d_in[8];
  const float*   bv   = (const float*)d_in[9];
  float* out = (float*)d_out;

  const size_t projElems = (size_t)B_DIM * SQ * DH;  // 16.7M elems, 33.5MB bf16
  bf16_t* Qp  = (bf16_t*)d_ws;
  bf16_t* Kp  = Qp + projElems;
  bf16_t* VpT = Kp + projElems;

  dim3 pg(DH / 128, (B_DIM * SQ) / 128);
  proj_kernel<0, 0><<<pg, 256, 0, stream>>>(q, Wq, bq, Qp,  B_DIM * SQ, DQK, DH);
  proj_kernel<0, 0><<<pg, 256, 0, stream>>>(k, Wk, bk, Kp,  B_DIM * SK, DQK, DH);
  proj_kernel<1, 1><<<pg, 256, 0, stream>>>(v, Wv, bv, VpT, B_DIM * SK, DQK, DV);

  dim3 fg(SQ / BQ, B_DIM);
  flash_kernel<<<fg, 256, 0, stream>>>(Qp, Kp, VpT, mask, out);
}
